// BinLinear_18270790877547
// MI455X (gfx1250) — matrix operand order repaired and checked by reading, not compiled, not run on hardware
//
#include <hip/hip_runtime.h>
#include <stdint.h>

// BitNet-style binary linear for MI455X (gfx1250, wave32, WMMA).
//
// out  = scale * (x @ sign(W)^T),  scale = mean|W|   -> d_out[0 .. 16M)
// assoc = sign(W)                                    -> d_out[16M .. 16M+1M)
//
// Roofline: GEMM = 34.4 GFLOP, min HBM traffic ~136 MB -> 5.8 us @ 23.3 TB/s.
// bf16 WMMA (16x16x32) is the balanced-precision choice; x is split into
// bf16 hi + bf16 residual (two WMMAs) so accumulation matches fp32 numerics
// (sign(W) = +-1 is exact in bf16).
//
// Workspace layout (needs ~2.1 MB):
//   [0      ..  1KB)  : 256 fp32 block partial abs-sums
//   [1KB    ..  1KB+4): fp32 scale
//   [4KB    ..  4KB+2MB): bf16 sign(W), same [N=1024][K=1024] layout as W

typedef __attribute__((ext_vector_type(16))) __bf16 v16bf;
typedef __attribute__((ext_vector_type(8)))  float  v8f;

#define DK      1024
#define DN      1024
#define MTOT    16384
#define BM      128
#define BN      128
#define BK      32
#define LDSROW  40      // padded LDS row stride in bf16 elems (80 B, conflict-free)

static __device__ __forceinline__ unsigned short f2bf(float f) {
  __bf16 h = (__bf16)f;
  return __builtin_bit_cast(unsigned short, h);
}
static __device__ __forceinline__ float bf2f(unsigned short u) {
  return (float)__builtin_bit_cast(__bf16, u);
}

// ---- scale = mean|W| : deterministic two-stage tree reduction -------------
__global__ void absmean_partial(const float* __restrict__ W,
                                float* __restrict__ partials) {
  __shared__ float red[256];
  const int t = threadIdx.x;
  const int base = blockIdx.x * 4096;
  float s = 0.f;
#pragma unroll
  for (int i = 0; i < 16; ++i) s += fabsf(W[base + t + i * 256]);
  red[t] = s;
  __syncthreads();
  for (int off = 128; off > 0; off >>= 1) {
    if (t < off) red[t] += red[t + off];
    __syncthreads();
  }
  if (t == 0) partials[blockIdx.x] = red[0];
}

__global__ void absmean_final(const float* __restrict__ partials,
                              float* __restrict__ scale) {
  __shared__ float red[256];
  const int t = threadIdx.x;
  red[t] = partials[t];
  __syncthreads();
  for (int off = 128; off > 0; off >>= 1) {
    if (t < off) red[t] += red[t + off];
    __syncthreads();
  }
  if (t == 0) *scale = red[0] * (1.0f / 1048576.0f);
}

// ---- binarize: assoc (fp32 sign) + bf16 +-1 weights in W's [N][K] layout --
__global__ void binarize(const float* __restrict__ W,
                         unsigned short* __restrict__ Wb,
                         float* __restrict__ assoc) {
  const int idx = (blockIdx.x * 256 + threadIdx.x) * 4;
  float4 w = *(const float4*)(W + idx);
  float4 s;
  s.x = (w.x > 0.f) ? 1.f : ((w.x < 0.f) ? -1.f : 0.f);
  s.y = (w.y > 0.f) ? 1.f : ((w.y < 0.f) ? -1.f : 0.f);
  s.z = (w.z > 0.f) ? 1.f : ((w.z < 0.f) ? -1.f : 0.f);
  s.w = (w.w > 0.f) ? 1.f : ((w.w < 0.f) ? -1.f : 0.f);
  *(float4*)(assoc + idx) = s;
  uint2 p;
  p.x = (uint32_t)f2bf(s.x) | ((uint32_t)f2bf(s.y) << 16);
  p.y = (uint32_t)f2bf(s.z) | ((uint32_t)f2bf(s.w) << 16);
  *(uint2*)(Wb + idx) = p;
}

// ---- main GEMM: 128x128 block tile, K-step 32, bf16 hi/lo split WMMA ------
__global__ __launch_bounds__(256) void bin_gemm(
    const float* __restrict__ X, const unsigned short* __restrict__ Wb,
    const float* __restrict__ scale_p, float* __restrict__ out) {
  __shared__ alignas(16) unsigned short As_hi[BM * LDSROW];
  __shared__ alignas(16) unsigned short As_lo[BM * LDSROW];
  __shared__ alignas(16) unsigned short Bs[BN * LDSROW];

  const int t = threadIdx.x;
  const int lane = t & 31;
  const int laneHalf = lane >> 4;   // 0 = lanes 0-15, 1 = lanes 16-31
  const int l16 = lane & 15;
  const int wave = t >> 5;          // 8 waves
  const int waveM = wave >> 2;      // 0..1 -> 64 rows each
  const int waveN = wave & 3;       // 0..3 -> 32 cols each
  const int blockN = blockIdx.x * BN;
  const int blockM = blockIdx.y * BM;

  v8f acc[4][2];
#pragma unroll
  for (int mi = 0; mi < 4; ++mi)
#pragma unroll
    for (int ni = 0; ni < 2; ++ni) acc[mi][ni] = {};

  const float* xg0 = X + (size_t)blockM * DK;
  const unsigned short* wg0 = Wb + (size_t)blockN * DK;

  for (int kt = 0; kt < DK / BK; ++kt) {
    // ---- stage A tile (128 x 32 fp32 -> bf16 hi/lo in LDS) ----
    const float* xg = xg0 + kt * BK;
#pragma unroll
    for (int i = 0; i < 4; ++i) {
      const int f = t + i * 256;       // float4 id 0..1023
      const int row = f >> 3;          // 0..127
      const int c4 = f & 7;            // float4 within 32-float row
      float4 v = *(const float4*)(xg + (size_t)row * DK + c4 * 4);
      const unsigned short h0 = f2bf(v.x), h1 = f2bf(v.y),
                           h2 = f2bf(v.z), h3 = f2bf(v.w);
      const unsigned short l0 = f2bf(v.x - bf2f(h0)), l1 = f2bf(v.y - bf2f(h1)),
                           l2 = f2bf(v.z - bf2f(h2)), l3 = f2bf(v.w - bf2f(h3));
      uint2 hp, lp;
      hp.x = (uint32_t)h0 | ((uint32_t)h1 << 16);
      hp.y = (uint32_t)h2 | ((uint32_t)h3 << 16);
      lp.x = (uint32_t)l0 | ((uint32_t)l1 << 16);
      lp.y = (uint32_t)l2 | ((uint32_t)l3 << 16);
      *(uint2*)&As_hi[row * LDSROW + c4 * 4] = hp;
      *(uint2*)&As_lo[row * LDSROW + c4 * 4] = lp;
    }
    // ---- stage B tile (128 x 32 bf16, already K-contiguous) ----
    const unsigned short* wg = wg0 + kt * BK;
#pragma unroll
    for (int i = 0; i < 2; ++i) {
      const int g = t + i * 256;       // uint4 id 0..511
      const int row = g >> 2;          // 0..127
      const int c = g & 3;             // 16B chunk within 64B row
      uint4 v = *(const uint4*)(wg + (size_t)row * DK + c * 8);
      *(uint4*)&Bs[row * LDSROW + c * 8] = v;
    }
    // prefetch next K-tile while we compute (global_prefetch_b8)
    if (kt + 1 < DK / BK) {
      __builtin_prefetch(xg + BK + (size_t)(t >> 1) * DK + (t & 1) * 16, 0, 0);
      __builtin_prefetch(wg + BK + (size_t)t * DK, 0, 0);
    }
    __syncthreads();

    // ---- load WMMA fragments from LDS (documented 16-bit layouts) ----
    // A 16x32: lanes 0-15 hold K0-7 & K16-23; lanes 16-31 hold K8-15 & K24-31.
    // B 32x16: lanes 0-15 hold K0-7 & K16-23 of col n; lanes 16-31 hold K8-15 & K24-31.
    union Frag { v16bf v; uint4 u[2]; };
    Frag ahi[4], alo[4], bfr[2];
#pragma unroll
    for (int mi = 0; mi < 4; ++mi) {
      const int row = waveM * 64 + mi * 16 + l16;
      const unsigned short* p = &As_hi[row * LDSROW] + laneHalf * 8;  // +16B
      ahi[mi].u[0] = *(const uint4*)p;
      ahi[mi].u[1] = *(const uint4*)(p + 16);                          // +32B
      const unsigned short* q = &As_lo[row * LDSROW] + laneHalf * 8;
      alo[mi].u[0] = *(const uint4*)q;
      alo[mi].u[1] = *(const uint4*)(q + 16);
    }
#pragma unroll
    for (int ni = 0; ni < 2; ++ni) {
      const int row = waveN * 32 + ni * 16 + l16;
      const unsigned short* p = &Bs[row * LDSROW] + laneHalf * 8;      // +16B
      bfr[ni].u[0] = *(const uint4*)p;
      bfr[ni].u[1] = *(const uint4*)(p + 16);                          // +32B
    }

    // ---- 16 WMMAs per wave per K-step (hi + lo accumulate into same C) ----
#pragma unroll
    for (int mi = 0; mi < 4; ++mi)
#pragma unroll
      for (int ni = 0; ni < 2; ++ni) {
        acc[mi][ni] = __builtin_amdgcn_wmma_f32_16x16x32_bf16(
            false, ahi[mi].v, false, bfr[ni].v, (short)0, acc[mi][ni],
            false, false);
        acc[mi][ni] = __builtin_amdgcn_wmma_f32_16x16x32_bf16(
            false, alo[mi].v, false, bfr[ni].v, (short)0, acc[mi][ni],
            false, false);
      }
    __syncthreads();
  }

  // ---- epilogue: scale and store (C layout: VGPR r -> M=r / M=r+8) ----
  const float scale = *scale_p;
#pragma unroll
  for (int mi = 0; mi < 4; ++mi)
#pragma unroll
    for (int ni = 0; ni < 2; ++ni) {
      const int m0 = blockM + waveM * 64 + mi * 16 + laneHalf * 8;
      const int n0 = blockN + waveN * 32 + ni * 16 + l16;
      float* o = out + (size_t)m0 * DN + n0;
#pragma unroll
      for (int r = 0; r < 8; ++r) o[(size_t)r * DN] = scale * acc[mi][ni][r];
    }
}

extern "C" void kernel_launch(void* const* d_in, const int* in_sizes, int n_in,
                              void* d_out, int out_size, void* d_ws, size_t ws_size,
                              hipStream_t stream) {
  const float* x = (const float*)d_in[0];   // [4,4096,1024] fp32
  const float* W = (const float*)d_in[1];   // [1024,1024] fp32
  float* out = (float*)d_out;                          // 16M fp32
  float* assoc = out + (size_t)MTOT * DN;              // + 1M fp32
  float* partials = (float*)d_ws;                      // 256 fp32
  float* scale = (float*)((char*)d_ws + 1024);         // 1 fp32
  unsigned short* Wb = (unsigned short*)((char*)d_ws + 4096);  // 1M bf16

  absmean_partial<<<256, 256, 0, stream>>>(W, partials);
  absmean_final<<<1, 256, 0, stream>>>(partials, scale);
  binarize<<<1024, 256, 0, stream>>>(W, Wb, assoc);
  dim3 grid(DN / BN, MTOT / BM);
  bin_gemm<<<grid, 256, 0, stream>>>(x, Wb, scale, out);
}
